// FusedMoEModularKernel_2886218023316
// MI455X (gfx1250) — compile-verified, matrix operand order checked by reading, not run on hardware
//
#include <hip/hip_runtime.h>
#include <stdint.h>

// Problem constants (match reference setup_inputs)
constexpr int T_   = 2048;   // tokens
constexpr int H_   = 1024;   // hidden
constexpr int I_   = 1408;   // intermediate
constexpr int E_   = 8;      // experts
constexpr int K_   = 2;      // top-k
constexpr int CAP_ = 1024;   // capacity
constexpr int S_   = T_ * K_; // 4096 flat slots

typedef __attribute__((ext_vector_type(16))) __bf16 v16bf;
typedef __attribute__((ext_vector_type(8)))  __bf16 v8bf;
typedef __attribute__((ext_vector_type(8)))  float  v8f;
typedef int gvec4i __attribute__((vector_size(16)));   // <4 x i32> for async builtin

static __device__ __forceinline__ unsigned short f2bf(float f) {
  union { float f; unsigned u; } v; v.f = f;
  unsigned r = v.u + 0x7FFFu + ((v.u >> 16) & 1u);   // round-to-nearest-even
  return (unsigned short)(r >> 16);
}

// ---------------------------------------------------------------------------
// Async global->LDS staging (CDNA5 GLOBAL_LOAD_ASYNC_TO_LDS_B128, ASYNCcnt).
// ---------------------------------------------------------------------------
#if __has_builtin(__builtin_amdgcn_global_load_async_to_lds_b128)
#define USE_ASYNC_COPY 1
#else
#define USE_ASYNC_COPY 0
#endif

static __device__ __forceinline__ void copy16_g2l(const unsigned short* g, unsigned short* l) {
#if USE_ASYNC_COPY
  __builtin_amdgcn_global_load_async_to_lds_b128(
      (__attribute__((address_space(1))) gvec4i*)(void*)(unsigned short*)g,
      (__attribute__((address_space(3))) gvec4i*)(void*)l, 0, 0);
#else
  *(uint4*)l = *(const uint4*)g;
#endif
}

static __device__ __forceinline__ void copy_wait() {
#if USE_ASYNC_COPY
#if __has_builtin(__builtin_amdgcn_s_wait_asynccnt)
  __builtin_amdgcn_s_wait_asynccnt(0);
#else
  asm volatile("s_wait_asynccnt 0x0" ::: "memory");
#endif
#endif
}

// Scheduling fence: keep fragment loads before it, WMMA chain after it.
static __device__ __forceinline__ void sched_fence() {
#if __has_builtin(__builtin_amdgcn_sched_barrier)
  __builtin_amdgcn_sched_barrier(0);
#endif
}

// ---------------------------------------------------------------------------
// 1. Routing: reproduce reference cumsum positions deterministically.
// ---------------------------------------------------------------------------
__global__ __launch_bounds__(256) void route_kernel(const int* __restrict__ ids,
                                                    int* __restrict__ pos,
                                                    int* __restrict__ keep) {
  __shared__ int cnt[256][E_];
  const int tid = threadIdx.x;
  #pragma unroll
  for (int e = 0; e < E_; ++e) cnt[tid][e] = 0;
  const int base = tid * 16;
  for (int j = 0; j < 16; ++j) cnt[tid][ids[base + j]]++;
  __syncthreads();
  if (tid < E_) {                 // exclusive prefix per expert over 256 threads
    int run = 0;
    for (int i = 0; i < 256; ++i) { int c = cnt[i][tid]; cnt[i][tid] = run; run += c; }
  }
  __syncthreads();
  for (int j = 0; j < 16; ++j) {  // replay in slot order
    const int s = base + j;
    const int id = ids[s];
    const int p = cnt[tid][id]++;
    pos[s]  = p;
    keep[s] = (p < CAP_) ? 1 : 0;
  }
}

// ---------------------------------------------------------------------------
// 2. f32 -> bf16 cast / zero fill
// ---------------------------------------------------------------------------
__global__ __launch_bounds__(256) void cast_bf16_kernel(const float* __restrict__ src,
                                                        unsigned short* __restrict__ dst,
                                                        long long n4) {
  for (long long i = (long long)blockIdx.x * blockDim.x + threadIdx.x; i < n4;
       i += (long long)gridDim.x * blockDim.x) {
    float4 v = ((const float4*)src)[i];
    uint2 o;
    o.x = (unsigned)f2bf(v.x) | ((unsigned)f2bf(v.y) << 16);
    o.y = (unsigned)f2bf(v.z) | ((unsigned)f2bf(v.w) << 16);
    ((uint2*)dst)[i] = o;
  }
}

__global__ __launch_bounds__(256) void zero_kernel(uint4* __restrict__ p, long long n) {
  const uint4 z = make_uint4(0u, 0u, 0u, 0u);
  for (long long i = (long long)blockIdx.x * blockDim.x + threadIdx.x; i < n;
       i += (long long)gridDim.x * blockDim.x) p[i] = z;
}

// ---------------------------------------------------------------------------
// 3. Scatter: X[id, pos, :] = bf16(hidden[token])
// ---------------------------------------------------------------------------
__global__ __launch_bounds__(256) void scatter_kernel(const float* __restrict__ hs,
                                                      const int* __restrict__ ids,
                                                      const int* __restrict__ pos,
                                                      const int* __restrict__ keep,
                                                      unsigned short* __restrict__ Xb) {
  const int s = blockIdx.x;
  if (!keep[s]) return;
  const int id = ids[s], p = pos[s], tk = s >> 1;
  const float4* src = (const float4*)(hs + (size_t)tk * H_);
  uint2* dst = (uint2*)(Xb + ((size_t)id * CAP_ + p) * H_);
  for (int i = threadIdx.x; i < H_ / 4; i += blockDim.x) {
    float4 v = src[i];
    uint2 o;
    o.x = (unsigned)f2bf(v.x) | ((unsigned)f2bf(v.y) << 16);
    o.y = (unsigned)f2bf(v.z) | ((unsigned)f2bf(v.w) << 16);
    dst[i] = o;
  }
}

// ---------------------------------------------------------------------------
// WMMA fragment load from an LDS tile with row stride 40 bf16.
// Lanes 0-15: K 0..7 & 16..23 of row=lane; lanes 16-31: K 8..15 & 24..31.
// ---------------------------------------------------------------------------
static __device__ __forceinline__ v16bf load_frag40(const unsigned short* base, int lane) {
  const int r  = lane & 15;
  const int kb = (lane >> 4) << 3;   // 0 or 8 elements
  const v8bf lo = *(const v8bf*)(base + r * 40 + kb);
  const v8bf hi = *(const v8bf*)(base + r * 40 + kb + 16);
  return __builtin_shufflevector(lo, hi, 0,1,2,3,4,5,6,7,8,9,10,11,12,13,14,15);
}

// ---------------------------------------------------------------------------
// 4. GEMM1 + SiLU:  act = silu(X@Wg^T) * (X@Wu^T)   M=1024 N=1408 K=1024
//    128x(64 gate + 64 up) tile per 256-thread block, double-buffered LDS,
//    async global->LDS staging overlapped with the WMMA chain.
// ---------------------------------------------------------------------------
__global__ __launch_bounds__(256) void gemm1_silu_kernel(const unsigned short* __restrict__ Xb,
                                                         const unsigned short* __restrict__ w1b,
                                                         unsigned short* __restrict__ actb) {
  __shared__ __align__(16) unsigned short As[2][128 * 40];
  __shared__ __align__(16) unsigned short Bg[2][64 * 40];
  __shared__ __align__(16) unsigned short Bu[2][64 * 40];

  const int n0 = blockIdx.x * 64;
  const int m0 = blockIdx.y * 128;
  const int e  = blockIdx.z;
  const int t  = threadIdx.x;
  const int lane = t & 31, wave = t >> 5;

  const unsigned short* Ap  = Xb  + ((size_t)e * CAP_ + m0) * H_;
  const unsigned short* Bgp = w1b + ((size_t)e * 2 * I_ + n0) * H_;
  const unsigned short* Bup = w1b + ((size_t)e * 2 * I_ + I_ + n0) * H_;

  v8f accg[4] = {}; v8f accu[4] = {};

  const int arow = t >> 1, ahalf = t & 1;   // A: 2 threads per row, 2x16B each
  const int brow = t >> 2, bq    = t & 3;   // B: 4 threads per row, 1x16B each

  auto stage = [&](int buf, int k0) {
    const unsigned short* aSrc = Ap + (size_t)arow * H_ + k0 + ahalf * 16;
    copy16_g2l(aSrc,     &As[buf][arow * 40 + ahalf * 16]);
    copy16_g2l(aSrc + 8, &As[buf][arow * 40 + ahalf * 16 + 8]);
    copy16_g2l(Bgp + (size_t)brow * H_ + k0 + bq * 8, &Bg[buf][brow * 40 + bq * 8]);
    copy16_g2l(Bup + (size_t)brow * H_ + k0 + bq * 8, &Bu[buf][brow * 40 + bq * 8]);
  };

  stage(0, 0);
  copy_wait();
  __syncthreads();

  int cur = 0;
  for (int k0 = 0; k0 < H_; k0 += 32) {
    if (k0 + 32 < H_) stage(cur ^ 1, k0 + 32);   // prefetch next tile (async)

    // Register-stage all fragments ...
    const v16bf a = load_frag40(&As[cur][wave * 16 * 40], lane);
    v16bf bg[4], bu[4];
    #pragma unroll
    for (int nt = 0; nt < 4; ++nt) {
      bg[nt] = load_frag40(&Bg[cur][nt * 16 * 40], lane);
      bu[nt] = load_frag40(&Bu[cur][nt * 16 * 40], lane);
    }
    sched_fence();   // ... then issue the WMMA chain back-to-back
    #pragma unroll
    for (int nt = 0; nt < 4; ++nt) {
      accg[nt] = __builtin_amdgcn_wmma_f32_16x16x32_bf16(false, a, false, bg[nt],
                                                         (short)0, accg[nt], false, false);
      accu[nt] = __builtin_amdgcn_wmma_f32_16x16x32_bf16(false, a, false, bu[nt],
                                                         (short)0, accu[nt], false, false);
    }
    copy_wait();        // this wave's async writes to buf[cur^1] have landed
    __syncthreads();    // ... and every other wave's too
    cur ^= 1;
  }

  // Epilogue: act = silu(gate) * up, bf16.  C/D layout: VGPR r -> M=r(+8 hi lanes), N=lane&15.
  const int mbase = m0 + wave * 16 + ((lane >> 4) << 3);
  const int nbase = n0 + (lane & 15);
  #pragma unroll
  for (int nt = 0; nt < 4; ++nt) {
    #pragma unroll
    for (int r = 0; r < 8; ++r) {
      const float g = accg[nt][r], u = accu[nt][r];
      const float sg = g / (1.0f + __expf(-g));
      actb[((size_t)e * CAP_ + (mbase + r)) * I_ + (nbase + nt * 16)] = f2bf(sg * u);
    }
  }
}

// ---------------------------------------------------------------------------
// 5. GEMM2: y = act @ w2^T,  M=1024 N=1024 K=1408, f32 out.
// ---------------------------------------------------------------------------
__global__ __launch_bounds__(256) void gemm2_kernel(const unsigned short* __restrict__ actb,
                                                    const unsigned short* __restrict__ w2b,
                                                    float* __restrict__ y) {
  __shared__ __align__(16) unsigned short As[2][128 * 40];
  __shared__ __align__(16) unsigned short Bs[2][64 * 40];

  const int n0 = blockIdx.x * 64;
  const int m0 = blockIdx.y * 128;
  const int e  = blockIdx.z;
  const int t  = threadIdx.x;
  const int lane = t & 31, wave = t >> 5;

  const unsigned short* Ap = actb + ((size_t)e * CAP_ + m0) * I_;
  const unsigned short* Bp = w2b  + ((size_t)e * H_ + n0) * I_;

  v8f acc[4] = {};

  const int arow = t >> 1, ahalf = t & 1;
  const int brow = t >> 2, bq    = t & 3;

  auto stage = [&](int buf, int k0) {
    const unsigned short* aSrc = Ap + (size_t)arow * I_ + k0 + ahalf * 16;
    copy16_g2l(aSrc,     &As[buf][arow * 40 + ahalf * 16]);
    copy16_g2l(aSrc + 8, &As[buf][arow * 40 + ahalf * 16 + 8]);
    copy16_g2l(Bp + (size_t)brow * I_ + k0 + bq * 8, &Bs[buf][brow * 40 + bq * 8]);
  };

  stage(0, 0);
  copy_wait();
  __syncthreads();

  int cur = 0;
  for (int k0 = 0; k0 < I_; k0 += 32) {
    if (k0 + 32 < I_) stage(cur ^ 1, k0 + 32);

    const v16bf a = load_frag40(&As[cur][wave * 16 * 40], lane);
    v16bf b[4];
    #pragma unroll
    for (int nt = 0; nt < 4; ++nt) b[nt] = load_frag40(&Bs[cur][nt * 16 * 40], lane);
    sched_fence();
    #pragma unroll
    for (int nt = 0; nt < 4; ++nt) {
      acc[nt] = __builtin_amdgcn_wmma_f32_16x16x32_bf16(false, a, false, b[nt],
                                                        (short)0, acc[nt], false, false);
    }
    copy_wait();
    __syncthreads();
    cur ^= 1;
  }

  const int mbase = m0 + wave * 16 + ((lane >> 4) << 3);
  const int nbase = n0 + (lane & 15);
  #pragma unroll
  for (int nt = 0; nt < 4; ++nt) {
    #pragma unroll
    for (int r = 0; r < 8; ++r) {
      y[((size_t)e * CAP_ + (mbase + r)) * H_ + (nbase + nt * 16)] = acc[nt][r];
    }
  }
}

// ---------------------------------------------------------------------------
// 6. Combine: out[t] = sum_k keep*w[t,k]*y[id_k, pos_k]
// ---------------------------------------------------------------------------
__global__ __launch_bounds__(256) void combine_kernel(const float* __restrict__ y,
                                                      const int* __restrict__ ids,
                                                      const int* __restrict__ pos,
                                                      const int* __restrict__ keep,
                                                      const float* __restrict__ tw,
                                                      float* __restrict__ out) {
  const int tk = blockIdx.x;
  const int s0 = 2 * tk, s1 = 2 * tk + 1;
  const int id0 = ids[s0], id1 = ids[s1];
  const int p0 = keep[s0] ? pos[s0] : 0;
  const int p1 = keep[s1] ? pos[s1] : 0;
  const float w0 = keep[s0] ? tw[s0] : 0.0f;
  const float w1 = keep[s1] ? tw[s1] : 0.0f;
  const float4* y0 = (const float4*)(y + ((size_t)id0 * CAP_ + p0) * H_);
  const float4* y1 = (const float4*)(y + ((size_t)id1 * CAP_ + p1) * H_);
  float4* o = (float4*)(out + (size_t)tk * H_);
  for (int i = threadIdx.x; i < H_ / 4; i += blockDim.x) {
    const float4 a = y0[i], b = y1[i];
    float4 r;
    r.x = w0 * a.x + w1 * b.x;
    r.y = w0 * a.y + w1 * b.y;
    r.z = w0 * a.z + w1 * b.z;
    r.w = w0 * a.w + w1 * b.w;
    o[i] = r;
  }
}

// ---------------------------------------------------------------------------
// Host launcher
// ---------------------------------------------------------------------------
extern "C" void kernel_launch(void* const* d_in, const int* in_sizes, int n_in,
                              void* d_out, int out_size, void* d_ws, size_t ws_size,
                              hipStream_t stream) {
  const float* hs  = (const float*)d_in[0];   // (T, H)
  const float* w1  = (const float*)d_in[1];   // (E, 2I, H)
  const float* w2  = (const float*)d_in[2];   // (E, H, I)
  const float* tw  = (const float*)d_in[3];   // (T, K)
  const int*   ids = (const int*)d_in[4];     // (T, K)
  float* out = (float*)d_out;                 // (T, H)

  // Workspace layout (bytes, all 16B aligned)
  char* w = (char*)d_ws;
  int* pos  = (int*)(w);                                  // 4096 * 4
  int* keep = (int*)(w + 16384);                          // 4096 * 4
  unsigned short* w1b  = (unsigned short*)(w + 32768);    // E*2I*H bf16
  unsigned short* w2b  = w1b + (size_t)E_ * 2 * I_ * H_;  // E*H*I bf16
  unsigned short* Xb   = w2b + (size_t)E_ * H_ * I_;      // E*CAP*H bf16
  unsigned short* actb = Xb  + (size_t)E_ * CAP_ * H_;    // E*CAP*I bf16
  float* y = (float*)(actb + (size_t)E_ * CAP_ * I_);     // E*CAP*H f32

  route_kernel<<<1, 256, 0, stream>>>(ids, pos, keep);
  cast_bf16_kernel<<<2048, 256, 0, stream>>>(w1, w1b, (long long)E_ * 2 * I_ * H_ / 4);
  cast_bf16_kernel<<<2048, 256, 0, stream>>>(w2, w2b, (long long)E_ * H_ * I_ / 4);
  zero_kernel<<<2048, 256, 0, stream>>>((uint4*)Xb, (long long)E_ * CAP_ * H_ * 2 / 16);
  scatter_kernel<<<S_, 256, 0, stream>>>(hs, ids, pos, keep, Xb);
  gemm1_silu_kernel<<<dim3(I_ / 64, CAP_ / 128, E_), 256, 0, stream>>>(Xb, w1b, actb);
  gemm2_kernel<<<dim3(H_ / 64, CAP_ / 128, E_), 256, 0, stream>>>(actb, w2b, y);
  combine_kernel<<<T_, 256, 0, stream>>>(y, ids, pos, keep, tw, out);
}
